// VoxelBackBone8xFocal_6708738916900
// MI455X (gfx1250) — compile-verified, matrix operand order checked by reading, not run on hardware
//
#include <hip/hip_runtime.h>

#ifndef __has_builtin
#define __has_builtin(x) 0
#endif
#if __has_builtin(__builtin_amdgcn_global_load_async_to_lds_b128)
#define USE_ASYNC_LDS 1
#else
#define USE_ASYNC_LDS 0
#endif

// ---------- types ----------
typedef __attribute__((ext_vector_type(16))) __bf16   v16bf;
typedef __attribute__((ext_vector_type(8)))  float    v8f;
typedef __attribute__((ext_vector_type(4)))  unsigned u32x4;
typedef __attribute__((ext_vector_type(4)))  int      v4i;

union AF { v16bf v; u32x4 u[2]; };

// ---------- conv params ----------
struct ConvP {
  const __bf16* in;     // [Di][Hi][Wpi][CIN] bf16, x-halo of 1 (index xi+1)
  const __bf16* wp;     // [ntaps][Cout][CIN] bf16
  const float*  gamma;
  const float*  beta;
  const float*  occ;    // [Do][Ho][Wo] float 0/1 at OUTPUT resolution
  __bf16*       out;    // [Do][Ho][Wpo][Cout] bf16 (x-halo), or null
  float*        outF32; // NCDHW fp32 (final layer) when out==null
  int Di, Hi, Wpi;
  int Do, Ho, Wo, Wpo;
  int Cout;
  int kd, kh, kw;
  int sz, sy, sx;
  int pz, py, px;
};

// ---------- implicit-GEMM conv via WMMA bf16, weights staged in LDS ----------
// grid: (ceil(nVoxTiles/8), Cout/16); block: 256 threads = 8 waves.
// Each block serves one 16-wide cout tile -> its weight slice
// [ntaps][16][CIN] (<= 27*16*64*2 = 55 KB) is staged once in LDS via
// global_load_async_to_lds_b128 (fallback: regular load + ds_store).
template <int CIN>
__global__ __launch_bounds__(256) void conv_wmma(ConvP p) {
  const int lane = threadIdx.x & 31;
  const int wave = threadIdx.x >> 5;
  const int tilesX = (p.Wo + 15) >> 4;
  const int coutBase = blockIdx.y << 4;
  const int n  = lane & 15;              // cout column for B/D, voxel row for A
  const int hi = lane >> 4;              // K-half selector
  const int ntaps = p.kd * p.kh * p.kw;

  __shared__ __bf16 wl[27 * 16 * CIN];

  // ---- cooperative weight staging (all waves; before any exit) ----
  {
    const int chunksPerTap = 2 * CIN;            // 16B chunks per (tap, 16 couts)
    const int nChunks = ntaps * chunksPerTap;
    for (int i = threadIdx.x; i < nChunks; i += blockDim.x) {
      int tap = i / chunksPerTap, rem = i - tap * chunksPerTap;
      const __bf16* src = p.wp + ((long)tap * p.Cout + coutBase) * CIN + rem * 8;
      __bf16* dst = &wl[tap * (16 * CIN) + rem * 8];
#if USE_ASYNC_LDS
      v4i* s0 = (v4i*)src;   // strip const, retype in generic AS
      v4i* d0 = (v4i*)dst;
      __builtin_amdgcn_global_load_async_to_lds_b128(
          (__attribute__((address_space(1))) v4i*)s0,
          (__attribute__((address_space(3))) v4i*)d0, 0, 0);
#else
      *(u32x4*)dst = *(const u32x4*)src;
#endif
    }
#if USE_ASYNC_LDS
    asm volatile("s_wait_asynccnt 0" ::: "memory");
#endif
    __syncthreads();
  }

  long vt = (long)blockIdx.x * (blockDim.x >> 5) + wave;
  long nVT = (long)tilesX * p.Ho * p.Do;
  if (vt >= nVT) return;                 // wave-uniform exit (EXEC full for WMMA)

  int tx = (int)(vt % tilesX); long t2 = vt / tilesX;
  int yo = (int)(t2 % p.Ho);
  int zo = (int)(t2 / p.Ho);

  const int xoA = (tx << 4) + n;         // this lane's A-row voxel
  const bool voxValid = (xoA < p.Wo);
  const int xi0 = xoA * p.sx - p.px;
  const int zi0 = zo * p.sz - p.pz;
  const int yi0 = yo * p.sy - p.py;
  const u32x4 z4 = {0u, 0u, 0u, 0u};

  v8f acc = {0.f, 0.f, 0.f, 0.f, 0.f, 0.f, 0.f, 0.f};

  if constexpr (CIN == 16) {
    // pack two taps per WMMA (K = 2 x 16 channels)
    for (int t = 0; t < ntaps; t += 2) {
      AF a, b;
#pragma unroll
      for (int e = 0; e < 2; ++e) {
        int tt = t + e;
        bool tv = tt < ntaps;
        int dz = 0, dy = 0, dx = 0;
        if (tv) { dx = tt % p.kw; int rr = tt / p.kw; dy = rr % p.kh; dz = rr / p.kh; }
        int zi = zi0 + dz, yi = yi0 + dy, xi = xi0 + dx;
        bool av = tv && voxValid &&
                  ((unsigned)zi < (unsigned)p.Di) && ((unsigned)yi < (unsigned)p.Hi);
        if (av)
          a.u[e] = *(const u32x4*)(p.in + (((long)zi * p.Hi + yi) * p.Wpi + (xi + 1)) * 16 + hi * 8);
        else
          a.u[e] = z4;
        if (tv)
          b.u[e] = *(const u32x4*)(&wl[(tt * 16 + n) * 16 + hi * 8]);
        else
          b.u[e] = z4;
      }
      acc = __builtin_amdgcn_wmma_f32_16x16x32_bf16(false, a.v, false, b.v,
                                                    (short)0, acc, false, false);
    }
  } else {
    int tap = 0;
    for (int dz = 0; dz < p.kd; ++dz) {
      int zi = zi0 + dz;
      bool zv = (unsigned)zi < (unsigned)p.Di;
      for (int dy = 0; dy < p.kh; ++dy) {
        int yi = yi0 + dy;
        bool yv = zv && ((unsigned)yi < (unsigned)p.Hi);
        for (int dx = 0; dx < p.kw; ++dx, ++tap) {
          int xi = xi0 + dx;
          bool av = yv && voxValid;
          const __bf16* abase = p.in + (((long)zi * p.Hi + yi) * p.Wpi + (xi + 1)) * CIN;
          const __bf16* wbase = &wl[(tap * 16 + n) * CIN];
#pragma unroll
          for (int s = 0; s < CIN / 32; ++s) {
            AF a, b;
            int c0 = s * 32 + hi * 8;
            if (av) {
              a.u[0] = *(const u32x4*)(abase + c0);
              a.u[1] = *(const u32x4*)(abase + c0 + 16);
            } else {
              a.u[0] = z4; a.u[1] = z4;
            }
            b.u[0] = *(const u32x4*)(wbase + c0);
            b.u[1] = *(const u32x4*)(wbase + c0 + 16);
            acc = __builtin_amdgcn_wmma_f32_16x16x32_bf16(false, a.v, false, b.v,
                                                          (short)0, acc, false, false);
          }
        }
      }
    }
  }

  // ---- fused BN + ReLU + occupancy mask + store (D layout: lane n = cout col,
  //      rows m = hi*8 + r in VGPR r) ----
  float g  = p.gamma[coutBase + n];
  float bb = p.beta[coutBase + n];
#pragma unroll
  for (int r = 0; r < 8; ++r) {
    int m  = hi * 8 + r;
    int xo = (tx << 4) + m;
    if (xo >= p.Wo) continue;
    float v = acc[r] * g + bb;
    v = v > 0.f ? v : 0.f;
    if (p.occ) v *= p.occ[((long)zo * p.Ho + yo) * p.Wo + xo];
    if (p.out)
      p.out[(((long)zo * p.Ho + yo) * p.Wpo + (xo + 1)) * p.Cout + coutBase + n] = (__bf16)v;
    else
      p.outF32[(((long)(coutBase + n) * p.Do + zo) * p.Ho + yo) * p.Wo + xo] = v;
  }
}

// ---------- utility kernels ----------
__global__ void zero_u4(u32x4* p, long n) {
  long i = (long)blockIdx.x * blockDim.x + threadIdx.x;
  u32x4 z = {0u, 0u, 0u, 0u};
  for (; i < n; i += (long)gridDim.x * blockDim.x) p[i] = z;
}

// scatter voxel features into 16-channel bf16 dense grid (ch 4..15 stay zero)
__global__ void scatter_k(const float* vf, const int* vc, __bf16* dense, float* occ,
                          int nv, int H, int W, int Wp) {
  int i = blockIdx.x * blockDim.x + threadIdx.x;
  if (i >= nv) return;
  int z = vc[i * 4 + 1], y = vc[i * 4 + 2], x = vc[i * 4 + 3];
  long base = (((long)z * H + y) * Wp + (x + 1)) * 16;
  for (int c = 0; c < 4; ++c) dense[base + c] = (__bf16)vf[i * 4 + c];
  occ[((long)z * H + y) * W + x] = 1.f;
}

// OIDHW fp32 -> [tap][O][CINP] bf16 (cin zero-padded to CINP)
__global__ void prep_w(const float* w, __bf16* wp, int O, int I,
                       int KD, int KH, int KW, int CINP) {
  long ntot = (long)KD * KH * KW * O * CINP;
  long i = (long)blockIdx.x * blockDim.x + threadIdx.x;
  if (i >= ntot) return;
  int ci = (int)(i % CINP); long r = i / CINP;
  int o = (int)(r % O); int tap = (int)(r / O);
  int dx = tap % KW; int rr = tap / KW; int dy = rr % KH; int dz = rr / KH;
  float v = (ci < I)
      ? w[(((long)o * I + ci) * KD + dz) * (KH * KW) + (long)dy * KW + dx]
      : 0.f;
  wp[i] = (__bf16)v;
}

// occupancy downsample: any-occupied within kernel window
__global__ void occ_down_k(const float* oi, float* oo,
                           int Di, int Hi, int Wi, int Do, int Ho, int Wo,
                           int kd, int kh, int kw, int sz, int sy, int sx,
                           int pz, int py, int px) {
  long n = (long)Do * Ho * Wo;
  long i = (long)blockIdx.x * blockDim.x + threadIdx.x;
  if (i >= n) return;
  int x = (int)(i % Wo); long r = i / Wo;
  int y = (int)(r % Ho); int z = (int)(r / Ho);
  float s = 0.f;
  for (int dz = 0; dz < kd; ++dz) {
    int zi = z * sz - pz + dz; if ((unsigned)zi >= (unsigned)Di) continue;
    for (int dy = 0; dy < kh; ++dy) {
      int yi = y * sy - py + dy; if ((unsigned)yi >= (unsigned)Hi) continue;
      for (int dx = 0; dx < kw; ++dx) {
        int xi = x * sx - px + dx; if ((unsigned)xi >= (unsigned)Wi) continue;
        s += oi[((long)zi * Hi + yi) * Wi + xi];
      }
    }
  }
  oo[i] = s > 0.f ? 1.f : 0.f;
}

// importance conv (Cout=1) + sigmoid + occupancy mask
__global__ void imp_k(const __bf16* in, const float* w, const float* occ, float* imp,
                      int C, int D, int H, int W, int Wp) {
  long nvox = (long)D * H * W;
  long i = (long)blockIdx.x * blockDim.x + threadIdx.x;
  if (i >= nvox) return;
  int x = (int)(i % W); long r = i / W;
  int y = (int)(r % H); int z = (int)(r / H);
  float acc = 0.f;
  for (int dz = 0; dz < 3; ++dz) {
    int zi = z + dz - 1; if ((unsigned)zi >= (unsigned)D) continue;
    for (int dy = 0; dy < 3; ++dy) {
      int yi = y + dy - 1; if ((unsigned)yi >= (unsigned)H) continue;
      for (int dx = 0; dx < 3; ++dx) {
        int xi = x + dx - 1;  // x-halo holds zeros
        const __bf16* b = in + (((long)zi * H + yi) * Wp + (xi + 1)) * C;
        int tap = (dz * 3 + dy) * 3 + dx;
        for (int c = 0; c < C; ++c) acc += (float)b[c] * w[c * 27 + tap];
      }
    }
  }
  float s = 1.f / (1.f + __expf(-acc));
  imp[i] = s * occ[i];
}

// -------- top-K(3072) threshold via 2-pass positive-float radix histogram -----
__global__ void hist_p1(const float* imp, long n, unsigned* h) {
  long i = (long)blockIdx.x * blockDim.x + threadIdx.x;
  if (i >= n) return;
  float v = imp[i];
  if (v > 0.f) {
    unsigned u = __float_as_uint(v) >> 20;
    if (u > 1023u) u = 1023u;
    atomicAdd(&h[u], 1u);
  }
}
__global__ void select1(const unsigned* h, unsigned* res, int K) {
  if (threadIdx.x != 0 || blockIdx.x != 0) return;
  unsigned cum = 0;
  for (int b = 1023; b >= 0; --b) {
    unsigned c = h[b];
    if (cum + c >= (unsigned)K) { res[0] = (unsigned)b; res[1] = (unsigned)K - cum; return; }
    cum += c;
  }
  res[0] = 0u; res[1] = 0xFFFFFFFFu;  // fewer than K positives -> thr = 0
}
__global__ void hist_p2(const float* imp, long n, const unsigned* res, unsigned* h2) {
  if (res[1] == 0xFFFFFFFFu) return;
  unsigned bin = res[0];
  long i = (long)blockIdx.x * blockDim.x + threadIdx.x;
  if (i >= n) return;
  float v = imp[i];
  if (v > 0.f) {
    unsigned u = __float_as_uint(v);
    if ((u >> 20) == bin) atomicAdd(&h2[(u >> 8) & 0xFFFu], 1u);
  }
}
__global__ void select2(const unsigned* h2, const unsigned* res, float* thr) {
  if (threadIdx.x != 0 || blockIdx.x != 0) return;
  if (res[1] == 0xFFFFFFFFu) { *thr = 0.f; return; }
  unsigned K = res[1], cum = 0;
  for (int s = 4095; s >= 0; --s) {
    unsigned c = h2[s];
    if (cum + c >= K) { *thr = __uint_as_float((res[0] << 20) | ((unsigned)s << 8)); return; }
    cum += c;
  }
  *thr = __uint_as_float(res[0] << 20);
}

// sel mask + reweighted activations xm = x * (sel ? imp : 1)
__global__ void focal_apply(const __bf16* x, const float* imp, const float* thrP,
                            float* sel, __bf16* xm, int C, int D, int H, int W, int Wp) {
  long nvox = (long)D * H * W;
  long i = (long)blockIdx.x * blockDim.x + threadIdx.x;
  if (i >= nvox) return;
  int xc = (int)(i % W); long r = i / W;
  int y = (int)(r % H); int z = (int)(r / H);
  float v = imp[i];
  float thr = *thrP;
  bool s = (v >= thr) && (v > 0.f);
  sel[i] = s ? 1.f : 0.f;
  float mult = s ? v : 1.f;
  long hidx = (((long)z * H + y) * Wp + (xc + 1)) * C;
  for (int c = 0; c < C; ++c) xm[hidx + c] = (__bf16)((float)x[hidx + c] * mult);
}

// occ = max(occ, dilate3x3x3(sel))  (in place)
__global__ void focal_dilate(const float* sel, float* occ, int D, int H, int W) {
  long nvox = (long)D * H * W;
  long i = (long)blockIdx.x * blockDim.x + threadIdx.x;
  if (i >= nvox) return;
  int x = (int)(i % W); long r = i / W;
  int y = (int)(r % H); int z = (int)(r / H);
  float s = 0.f;
  for (int dz = -1; dz <= 1; ++dz) {
    int zi = z + dz; if ((unsigned)zi >= (unsigned)D) continue;
    for (int dy = -1; dy <= 1; ++dy) {
      int yi = y + dy; if ((unsigned)yi >= (unsigned)H) continue;
      for (int dx = -1; dx <= 1; ++dx) {
        int xi = x + dx; if ((unsigned)xi >= (unsigned)W) continue;
        s += sel[((long)zi * H + yi) * W + xi];
      }
    }
  }
  if (s > 0.f) occ[i] = 1.f;
}

// ---------- host side ----------
struct Dims { int D, H, W, Wp; };

static void conv(int CIN, const void* in, const void* wp, const float* g, const float* b,
                 const float* occ, void* out, float* outF32, Dims di, Dims dout, int Cout,
                 int kd, int kh, int kw, int sz, int sy, int sx, int pz, int py, int px,
                 hipStream_t s) {
  ConvP p;
  p.in = (const __bf16*)in;  p.wp = (const __bf16*)wp;
  p.gamma = g; p.beta = b; p.occ = occ;
  p.out = (__bf16*)out; p.outF32 = outF32;
  p.Di = di.D; p.Hi = di.H; p.Wpi = di.Wp;
  p.Do = dout.D; p.Ho = dout.H; p.Wo = dout.W; p.Wpo = dout.Wp;
  p.Cout = Cout; p.kd = kd; p.kh = kh; p.kw = kw;
  p.sz = sz; p.sy = sy; p.sx = sx; p.pz = pz; p.py = py; p.px = px;
  long tilesX = (dout.W + 15) / 16;
  long nVT = tilesX * dout.H * dout.D;
  dim3 grid((unsigned)((nVT + 7) / 8), (unsigned)(Cout / 16));
  if (CIN == 16)      conv_wmma<16><<<grid, 256, 0, s>>>(p);
  else if (CIN == 32) conv_wmma<32><<<grid, 256, 0, s>>>(p);
  else                conv_wmma<64><<<grid, 256, 0, s>>>(p);
}

static void prep(const float* w, void* wp, int O, int I, int KD, int KH, int KW,
                 int CINP, hipStream_t s) {
  long n = (long)KD * KH * KW * O * CINP;
  prep_w<<<(unsigned)((n + 255) / 256), 256, 0, s>>>(w, (__bf16*)wp, O, I, KD, KH, KW, CINP);
}

static void occ_down(const float* oi, float* oo, Dims di, Dims dout,
                     int kd, int kh, int kw, int sz, int sy, int sx,
                     int pz, int py, int px, hipStream_t s) {
  long n = (long)dout.D * dout.H * dout.W;
  occ_down_k<<<(unsigned)((n + 255) / 256), 256, 0, s>>>(
      oi, oo, di.D, di.H, di.W, dout.D, dout.H, dout.W, kd, kh, kw, sz, sy, sx, pz, py, px);
}

static void focal(const void* xin, void* xm, const float* w_imp, float* occ,
                  float* imp, float* sel, unsigned* h1, unsigned* h2, unsigned* res,
                  float* thr, int C, Dims d, hipStream_t s) {
  long nvox = (long)d.D * d.H * d.W;
  unsigned blk = (unsigned)((nvox + 255) / 256);
  imp_k<<<blk, 256, 0, s>>>((const __bf16*)xin, w_imp, occ, imp, C, d.D, d.H, d.W, d.Wp);
  hist_p1<<<blk, 256, 0, s>>>(imp, nvox, h1);
  select1<<<1, 32, 0, s>>>(h1, res, 3072);
  hist_p2<<<blk, 256, 0, s>>>(imp, nvox, res, h2);
  select2<<<1, 32, 0, s>>>(h2, res, thr);
  focal_apply<<<blk, 256, 0, s>>>((const __bf16*)xin, imp, thr, sel, (__bf16*)xm,
                                  C, d.D, d.H, d.W, d.Wp);
  focal_dilate<<<blk, 256, 0, s>>>(sel, occ, d.D, d.H, d.W);
}

extern "C" void kernel_launch(void* const* d_in, const int* in_sizes, int n_in,
                              void* d_out, int out_size, void* d_ws, size_t ws_size,
                              hipStream_t stream) {
  (void)n_in; (void)out_size; (void)ws_size;
  const float* vf = (const float*)d_in[0];
  const int*   vc = (const int*)d_in[1];
  const int NV = in_sizes[0] / 4;
#define PRM(k) ((const float*)d_in[2 + (k)])
  // param order: 0 w_in 1 g_in 2 b_in 3 w1a 4 g1a 5 b1a 6 w1_imp 7 w1f 8 g1f 9 b1f
  // 10 w2a 11 g2a 12 b2a 13 w2b 14 g2b 15 b2b 16 w2c 17 g2c 18 b2c 19 w2_imp
  // 20 w2f 21 g2f 22 b2f 23 w3a 24 g3a 25 b3a 26 w3b 27 g3b 28 b3b 29 w3c 30 g3c 31 b3c
  // 32 w3_imp 33 w3f 34 g3f 35 b3f 36 w4a 37 g4a 38 b4a 39 w4b 40 g4b 41 b4b
  // 42 w4c 43 g4c 44 b4c 45 w_out 46 g_out 47 b_out

  const Dims d1{41, 160, 160, 162}, d2{21, 80, 80, 82}, d3{11, 40, 40, 42},
             d4{5, 20, 20, 22}, dO{2, 20, 20, 22};
  const size_t S1 = (size_t)41 * 160 * 162 * 16 * 2;   // 34,007,040
  const size_t S2 = (size_t)21 * 80 * 82 * 32 * 2;     //  8,816,640
  const size_t S3 = (size_t)11 * 40 * 42 * 64 * 2;     //  2,365,440
  const size_t S4 = (size_t)5 * 20 * 22 * 64 * 2;      //    281,600
  const long N1 = 41L * 160 * 160, N2 = 21L * 80 * 80, N3 = 11L * 40 * 40,
             N4 = 5L * 20 * 20,    NO = 2L * 20 * 20;

  char* ws = (char*)d_ws;
  size_t off = 0;
  auto alloc = [&](size_t bytes) -> void* {
    off = (off + 255) & ~(size_t)255;
    void* pp = ws + off; off += bytes; return pp;
  };
  auto a256 = [](size_t x) { return (x + 255) & ~(size_t)255; };

  void* bufA = alloc(S1);                   // dense0 / x1a / x1 ; stage3+4 reuse
  void* bufB = alloc(S1);                   // x0 / xm1 ; stage2 reuse
  float* occ0 = (float*)alloc(N1 * 4);
  float* imp1 = (float*)alloc(N1 * 4);
  float* sel1 = (float*)alloc(N1 * 4);
  float* occ2 = (float*)alloc(N2 * 4);
  float* imp2 = (float*)alloc(N2 * 4);
  float* sel2 = (float*)alloc(N2 * 4);
  float* occ3 = (float*)alloc(N3 * 4);
  float* imp3 = (float*)alloc(N3 * 4);
  float* sel3 = (float*)alloc(N3 * 4);
  float* occ4 = (float*)alloc(N4 * 4);
  float* occO = (float*)alloc(NO * 4);
  // prepacked weights
  void* wpIn = alloc(27 * 16 * 16 * 2);
  void* wp1a = alloc(27 * 16 * 16 * 2);
  void* wp1f = alloc(27 * 16 * 16 * 2);
  void* wp2a = alloc(27 * 32 * 16 * 2);
  void* wp2b = alloc(27 * 32 * 32 * 2);
  void* wp2c = alloc(27 * 32 * 32 * 2);
  void* wp2f = alloc(27 * 32 * 32 * 2);
  void* wp3a = alloc(27 * 64 * 32 * 2);
  void* wp3b = alloc(27 * 64 * 64 * 2);
  void* wp3c = alloc(27 * 64 * 64 * 2);
  void* wp3f = alloc(27 * 64 * 64 * 2);
  void* wp4a = alloc(27 * 64 * 64 * 2);
  void* wp4b = alloc(27 * 64 * 64 * 2);
  void* wp4c = alloc(27 * 64 * 64 * 2);
  void* wpOut = alloc(3 * 128 * 64 * 2);
  // top-k scratch (per stage)
  unsigned* h1a = (unsigned*)alloc(1024 * 4); unsigned* h2a = (unsigned*)alloc(4096 * 4);
  unsigned* ra  = (unsigned*)alloc(2 * 4);    float*    ta  = (float*)alloc(4);
  unsigned* h1b = (unsigned*)alloc(1024 * 4); unsigned* h2b = (unsigned*)alloc(4096 * 4);
  unsigned* rb  = (unsigned*)alloc(2 * 4);    float*    tb  = (float*)alloc(4);
  unsigned* h1c = (unsigned*)alloc(1024 * 4); unsigned* h2c = (unsigned*)alloc(4096 * 4);
  unsigned* rc  = (unsigned*)alloc(2 * 4);    float*    tc  = (float*)alloc(4);
  const size_t arena = off;

  // aliased stage buffers (placed in regions that are dead by the time they
  // are first written; explicitly re-zeroed before use so halos stay clean)
  void* bufC = bufB;                                   // stage2 ping
  void* bufD = (char*)bufB + a256(S2);                 // stage2 pong
  void* bufE = bufA;                                   // stage3 ping
  void* bufF = (char*)bufA + a256(S3);                 // stage3 pong
  void* bufG = (char*)bufA + 2 * a256(S3);             // stage4 ping
  void* bufH = (char*)bufG + a256(S4);                 // stage4 pong

  // ---- init: zero full arena (deterministic across graph replays) ----
  {
    long n16 = (long)((arena + 15) / 16);
    unsigned blk = (unsigned)((n16 + 255) / 256);
    if (blk > 16384u) blk = 16384u;
    zero_u4<<<blk, 256, 0, stream>>>((u32x4*)ws, n16);
  }

  // ---- weight prepack ----
  prep(PRM(0),  wpIn, 16,  4, 3, 3, 3, 16, stream);
  prep(PRM(3),  wp1a, 16, 16, 3, 3, 3, 16, stream);
  prep(PRM(7),  wp1f, 16, 16, 3, 3, 3, 16, stream);
  prep(PRM(10), wp2a, 32, 16, 3, 3, 3, 16, stream);
  prep(PRM(13), wp2b, 32, 32, 3, 3, 3, 32, stream);
  prep(PRM(16), wp2c, 32, 32, 3, 3, 3, 32, stream);
  prep(PRM(20), wp2f, 32, 32, 3, 3, 3, 32, stream);
  prep(PRM(23), wp3a, 64, 32, 3, 3, 3, 32, stream);
  prep(PRM(26), wp3b, 64, 64, 3, 3, 3, 64, stream);
  prep(PRM(29), wp3c, 64, 64, 3, 3, 3, 64, stream);
  prep(PRM(33), wp3f, 64, 64, 3, 3, 3, 64, stream);
  prep(PRM(36), wp4a, 64, 64, 3, 3, 3, 64, stream);
  prep(PRM(39), wp4b, 64, 64, 3, 3, 3, 64, stream);
  prep(PRM(42), wp4c, 64, 64, 3, 3, 3, 64, stream);
  prep(PRM(45), wpOut, 128, 64, 3, 1, 1, 64, stream);

  // ---- scatter ----
  scatter_k<<<(NV + 255) / 256, 256, 0, stream>>>(vf, vc, (__bf16*)bufA, occ0,
                                                  NV, d1.H, d1.W, d1.Wp);

  // ---- stage 1 (C=16, 41x160x160) ----
  conv(16, bufA, wpIn, PRM(1), PRM(2), occ0, bufB, nullptr, d1, d1, 16,
       3, 3, 3, 1, 1, 1, 1, 1, 1, stream);                               // x0
  conv(16, bufB, wp1a, PRM(4), PRM(5), occ0, bufA, nullptr, d1, d1, 16,
       3, 3, 3, 1, 1, 1, 1, 1, 1, stream);                               // x1a
  focal(bufA, bufB, PRM(6), occ0, imp1, sel1, h1a, h2a, ra, ta, 16, d1, stream);
  conv(16, bufB, wp1f, PRM(8), PRM(9), occ0, bufA, nullptr, d1, d1, 16,
       3, 3, 3, 1, 1, 1, 1, 1, 1, stream);                               // x1

  // ---- stage 2 (C=32, 21x80x80) ----
  occ_down(occ0, occ2, d1, d2, 3, 3, 3, 2, 2, 2, 1, 1, 1, stream);
  { // zero stage-2 region (aliases dead xm1 in bufB) so output halos are clean
    long n16 = (long)((2 * a256(S2) + 15) / 16);
    zero_u4<<<(unsigned)((n16 + 255) / 256), 256, 0, stream>>>((u32x4*)bufC, n16);
  }
  conv(16, bufA, wp2a, PRM(11), PRM(12), occ2, bufC, nullptr, d1, d2, 32,
       3, 3, 3, 2, 2, 2, 1, 1, 1, stream);                               // x2a
  conv(32, bufC, wp2b, PRM(14), PRM(15), occ2, bufD, nullptr, d2, d2, 32,
       3, 3, 3, 1, 1, 1, 1, 1, 1, stream);                               // x2b
  conv(32, bufD, wp2c, PRM(17), PRM(18), occ2, bufC, nullptr, d2, d2, 32,
       3, 3, 3, 1, 1, 1, 1, 1, 1, stream);                               // x2c
  focal(bufC, bufD, PRM(19), occ2, imp2, sel2, h1b, h2b, rb, tb, 32, d2, stream);
  conv(32, bufD, wp2f, PRM(21), PRM(22), occ2, bufC, nullptr, d2, d2, 32,
       3, 3, 3, 1, 1, 1, 1, 1, 1, stream);                               // x2

  // ---- stage 3 (C=64, 11x40x40) ----
  occ_down(occ2, occ3, d2, d3, 3, 3, 3, 2, 2, 2, 1, 1, 1, stream);
  { // zero stage-3/4 region (aliases dead x1 in bufA)
    long n16 = (long)((2 * a256(S3) + 2 * a256(S4) + 15) / 16);
    zero_u4<<<(unsigned)((n16 + 255) / 256), 256, 0, stream>>>((u32x4*)bufE, n16);
  }
  conv(32, bufC, wp3a, PRM(24), PRM(25), occ3, bufE, nullptr, d2, d3, 64,
       3, 3, 3, 2, 2, 2, 1, 1, 1, stream);                               // x3a
  conv(64, bufE, wp3b, PRM(27), PRM(28), occ3, bufF, nullptr, d3, d3, 64,
       3, 3, 3, 1, 1, 1, 1, 1, 1, stream);                               // x3b
  conv(64, bufF, wp3c, PRM(30), PRM(31), occ3, bufE, nullptr, d3, d3, 64,
       3, 3, 3, 1, 1, 1, 1, 1, 1, stream);                               // x3c
  focal(bufE, bufF, PRM(32), occ3, imp3, sel3, h1c, h2c, rc, tc, 64, d3, stream);
  conv(64, bufF, wp3f, PRM(34), PRM(35), occ3, bufE, nullptr, d3, d3, 64,
       3, 3, 3, 1, 1, 1, 1, 1, 1, stream);                               // x3

  // ---- stage 4 (C=64, 5x20x20; z pad 0) ----
  occ_down(occ3, occ4, d3, d4, 3, 3, 3, 2, 2, 2, 0, 1, 1, stream);
  conv(64, bufE, wp4a, PRM(37), PRM(38), occ4, bufG, nullptr, d3, d4, 64,
       3, 3, 3, 2, 2, 2, 0, 1, 1, stream);                               // x4a
  conv(64, bufG, wp4b, PRM(40), PRM(41), occ4, bufH, nullptr, d4, d4, 64,
       3, 3, 3, 1, 1, 1, 1, 1, 1, stream);                               // x4b
  conv(64, bufH, wp4c, PRM(43), PRM(44), occ4, bufG, nullptr, d4, d4, 64,
       3, 3, 3, 1, 1, 1, 1, 1, 1, stream);                               // x4c

  // ---- output conv (3,1,1) stride (2,1,1), 128 ch, fp32 NCDHW to d_out ----
  occ_down(occ4, occO, d4, dO, 3, 1, 1, 2, 1, 1, 0, 0, 0, stream);
  conv(64, bufG, wpOut, PRM(46), PRM(47), occO, nullptr, (float*)d_out, d4, dO, 128,
       3, 1, 1, 2, 1, 1, 0, 0, 0, stream);
#undef PRM
}
